// GATWithSingleHead_52630529245191
// MI455X (gfx1250) — compile-verified, matrix operand order checked by reading, not run on hardware
//
#include <hip/hip_runtime.h>
#include <hip/hip_bf16.h>

#define HEADS 4
#define FEAT  256
#define KDIM  256          // K of both GEMMs (fin and z-dim); fixed
#define LEAKY 0.2f

typedef __attribute__((ext_vector_type(16))) __bf16        v16bf;
typedef __attribute__((ext_vector_type(8)))  float         v8f;
typedef __attribute__((ext_vector_type(4)))  unsigned int  v4u;

union FragU { v4u q[2]; v16bf v; };

__device__ __forceinline__ unsigned short f2bf(float f) {
    unsigned int u = __float_as_uint(f);
    unsigned int r = u + 0x7FFFu + ((u >> 16) & 1u);   // round-to-nearest-even
    return (unsigned short)(r >> 16);
}

// ---------- casts / packing ----------
__global__ void k_cast_bf16(const float* __restrict__ in, unsigned short* __restrict__ out, int n) {
    int i = blockIdx.x * 256 + threadIdx.x;
    if (i < n) out[i] = f2bf(in[i]);
}

// in[rows][cols] (f32) -> out[cols][rows] (bf16)
__global__ void k_transpose_bf16(const float* __restrict__ in, unsigned short* __restrict__ out,
                                 int rows, int cols) {
    int i = blockIdx.x * 256 + threadIdx.x;
    if (i < rows * cols) {
        int r = i / cols, c = i % cols;
        out[(size_t)c * rows + r] = f2bf(in[i]);
    }
}

// ---------- WMMA GEMM: C[M,N] = A[M,KDIM] * Bt[N,KDIM]^T  (bf16 in, f32 out) ----------
// act: 0 = none, 1 = sigmoid. Out-of-range rows/cols are CLAMPED on load; the
// store guard uses the real indices, so no predication in the inner loop.
// Fixed base pointers + fully unrolled, double-buffered K loop; min-blocks=1
// gives the register allocator room to keep both fragment buffers live so the
// next step's 10 b128 loads overlap the current step's 4 WMMAs.
__global__ __launch_bounds__(256, 1)
void k_wmma_gemm(const unsigned short* __restrict__ A, const unsigned short* __restrict__ Bt,
                 float* __restrict__ C, int M, int N, int act) {
    const int lane = threadIdx.x & 31;
    const int wave = threadIdx.x >> 5;
    const int wr = wave >> 1, wc = wave & 1;
    const int rowBase = blockIdx.y * 64 + wr * 16;
    const int colBase = blockIdx.x * 128 + wc * 64;
    const int g   = lane >> 4;    // half-wave group: K offset +8
    const int l16 = lane & 15;

    v8f acc[4];
    #pragma unroll
    for (int t = 0; t < 4; t++)
        #pragma unroll
        for (int j = 0; j < 8; j++) acc[t][j] = 0.f;

    // fixed base pointers (lane-dependent, loop-invariant)
    const int rowA = min(rowBase + l16, M - 1);
    const unsigned short* __restrict__ ap = A + (size_t)rowA * KDIM + 8 * g;
    const unsigned short* __restrict__ bp[4];
    #pragma unroll
    for (int t = 0; t < 4; t++) {
        int colB = min(colBase + t * 16 + l16, N - 1);
        bp[t] = Bt + (size_t)colB * KDIM + 8 * g;
    }

    // double-buffered fragments
    FragU aF[2];
    FragU bF[2][4];
    auto load_step = [&](int buf, int k) {
        aF[buf].q[0] = *(const v4u*)(ap + k);
        aF[buf].q[1] = *(const v4u*)(ap + k + 16);
        #pragma unroll
        for (int t = 0; t < 4; t++) {
            bF[buf][t].q[0] = *(const v4u*)(bp[t] + k);
            bF[buf][t].q[1] = *(const v4u*)(bp[t] + k + 16);
        }
    };

    load_step(0, 0);
    #pragma unroll
    for (int k = 0; k < KDIM; k += 32) {
        const int cur = (k >> 5) & 1, nxt = cur ^ 1;
        if (k + 32 < KDIM) load_step(nxt, k + 32);   // prefetch next step
        #pragma unroll
        for (int t = 0; t < 4; t++)
            acc[t] = __builtin_amdgcn_wmma_f32_16x16x32_bf16(
                false, aF[cur].v, false, bF[cur][t].v, (short)0, acc[t], false, false);
    }

    // C/D layout: lane -> col = l16, VGPR r -> row = r + 8*g
    #pragma unroll
    for (int t = 0; t < 4; t++) {
        int n = colBase + t * 16 + l16;
        if (n >= N) continue;
        #pragma unroll
        for (int r = 0; r < 8; r++) {
            int m = rowBase + r + 8 * g;
            if (m < M) {
                float v = acc[t][r];
                if (act == 1) v = 1.f / (1.f + __expf(-v));
                C[(size_t)m * N + n] = v;
            }
        }
    }
}

// ---------- attention logits: one wave per (node, head) ----------
__global__ void k_logits(const float* __restrict__ h, const float* __restrict__ a_src,
                         const float* __restrict__ a_dst, float* __restrict__ al_s,
                         float* __restrict__ al_d, int Nn) {
    int wid  = (int)((blockIdx.x * blockDim.x + threadIdx.x) >> 5);
    int lane = threadIdx.x & 31;
    if (wid >= Nn * HEADS) return;
    int node = wid >> 2, hd = wid & 3;
    const float* hp = h + (size_t)node * (HEADS * FEAT) + hd * FEAT;
    const float* as = a_src + hd * FEAT;
    const float* ad = a_dst + hd * FEAT;
    float ss = 0.f, sd = 0.f;
    for (int f = lane; f < FEAT; f += 32) {
        float v = hp[f];
        ss += v * as[f];
        sd += v * ad[f];
    }
    #pragma unroll
    for (int off = 16; off; off >>= 1) {
        ss += __shfl_down(ss, off, 32);
        sd += __shfl_down(sd, off, 32);
    }
    if (lane == 0) { al_s[wid] = ss; al_d[wid] = sd; }
}

// ---------- init scratch (m=-inf, s=0, agg=0) ----------
__global__ void k_init(float* __restrict__ m, float* __restrict__ s, float* __restrict__ agg,
                       int nh, int nf) {
    int i = blockIdx.x * 256 + threadIdx.x;
    if (i < nh) { m[i] = -1e30f; s[i] = 0.f; }
    if (i < nf) agg[i] = 0.f;
}

__device__ __forceinline__ void atomicMaxF(float* addr, float val) {
    unsigned int* ua = (unsigned int*)addr;
    unsigned int cur = __float_as_uint(*addr);
    while (__uint_as_float(cur) < val) {
        unsigned int prev = atomicCAS(ua, cur, __float_as_uint(val));
        if (prev == cur) break;
        cur = prev;
    }
}

__device__ __forceinline__ float edge_score(const float* al_s, const float* al_d,
                                            int s, int d, int hd) {
    float v = al_s[s * HEADS + hd] + al_d[d * HEADS + hd];
    return (v > 0.f) ? v : LEAKY * v;
}

// ---------- segment max over dst ----------
__global__ void k_edge_max(const int* __restrict__ ei, const float* __restrict__ al_s,
                           const float* __restrict__ al_d, float* __restrict__ m,
                           int E, int Etot) {
    int i = blockIdx.x * 256 + threadIdx.x;
    if (i >= Etot * HEADS) return;
    int e = i >> 2, hd = i & 3;
    int s = (e < E) ? ei[e]     : (e - E);
    int d = (e < E) ? ei[E + e] : (e - E);
    atomicMaxF(&m[d * HEADS + hd], edge_score(al_s, al_d, s, d, hd));
}

// ---------- exp(e - max) and segment sum ----------
__global__ void k_edge_exp(const int* __restrict__ ei, const float* __restrict__ al_s,
                           const float* __restrict__ al_d, const float* __restrict__ m,
                           float* __restrict__ ex, float* __restrict__ ssum, int E, int Etot) {
    int i = blockIdx.x * 256 + threadIdx.x;
    if (i >= Etot * HEADS) return;
    int e = i >> 2, hd = i & 3;
    int s = (e < E) ? ei[e]     : (e - E);
    int d = (e < E) ? ei[E + e] : (e - E);
    float xv = __expf(edge_score(al_s, al_d, s, d, hd) - m[d * HEADS + hd]);
    ex[i] = xv;
    atomicAdd(&ssum[d * HEADS + hd], xv);
}

// ---------- weighted aggregation: one wave per edge ----------
__global__ void k_aggregate(const int* __restrict__ ei, const float* __restrict__ h,
                            const float* __restrict__ ex, const float* __restrict__ ssum,
                            float* __restrict__ agg, int E, int Etot) {
    int wave = threadIdx.x >> 5, lane = threadIdx.x & 31;
    int e = blockIdx.x * 8 + wave;
    if (e >= Etot) return;
    int s = (e < E) ? ei[e]     : (e - E);
    int d = (e < E) ? ei[E + e] : (e - E);
    float alpha[HEADS];
    #pragma unroll
    for (int hd = 0; hd < HEADS; hd++)
        alpha[hd] = ex[e * HEADS + hd] / (ssum[d * HEADS + hd] + 1e-16f) * 0.25f; // head-mean folded
    const float* hs = h + (size_t)s * (HEADS * FEAT);
    float* ap = agg + (size_t)d * FEAT;
    for (int f = lane; f < FEAT; f += 32) {
        float v = 0.f;
        #pragma unroll
        for (int hd = 0; hd < HEADS; hd++) v += alpha[hd] * hs[hd * FEAT + f];
        atomicAdd(&ap[f], v);
    }
}

// ---------- bias + activation; emit f32 + bf16 copies ----------
__global__ void k_finalize(const float* __restrict__ agg, const float* __restrict__ bias,
                           float* __restrict__ outf, unsigned short* __restrict__ outbf,
                           int Nn, int mode /*0 relu, 1 tanh*/) {
    int i = blockIdx.x * 256 + threadIdx.x;
    if (i >= Nn * FEAT) return;
    int f = i & (FEAT - 1);
    float v = agg[i] + bias[f];
    v = mode ? tanhf(v) : fmaxf(v, 0.f);
    outf[i] = v;
    outbf[i] = f2bf(v);
}

extern "C" void kernel_launch(void* const* d_in, const int* in_sizes, int n_in,
                              void* d_out, int out_size, void* d_ws, size_t ws_size,
                              hipStream_t stream) {
    (void)n_in; (void)out_size; (void)ws_size;
    const float* x  = (const float*)d_in[0];
    const int*   ei = (const int*)d_in[1];
    const int Nn   = in_sizes[0] / FEAT;   // 10000
    const int E    = in_sizes[1] / 2;      // 320000
    const int Etot = E + Nn;               // self loops appended
    float* out = (float*)d_out;            // [Nn*Nn adj][Nn*FEAT z]

    char* ws = (char*)d_ws;
    auto alloc = [&](size_t bytes) -> char* {
        char* p = ws; ws += (bytes + 255) & ~(size_t)255; return p;
    };
    unsigned short* actbf = (unsigned short*)alloc((size_t)Nn * FEAT * 2);           // activations bf16 (also z)
    unsigned short* Wt    = (unsigned short*)alloc((size_t)HEADS * FEAT * FEAT * 2); // W^T bf16 [1024,256]
    float* hbuf  = (float*)alloc((size_t)Nn * HEADS * FEAT * 4);                     // [N,1024]
    float* al_s  = (float*)alloc((size_t)Nn * HEADS * 4);
    float* al_d  = (float*)alloc((size_t)Nn * HEADS * 4);
    float* mbuf  = (float*)alloc((size_t)Nn * HEADS * 4);
    float* sbuf  = (float*)alloc((size_t)Nn * HEADS * 4);
    float* exbuf = (float*)alloc((size_t)Etot * HEADS * 4);
    float* agg   = (float*)alloc((size_t)Nn * FEAT * 4);
    float* actf  = (float*)alloc((size_t)Nn * FEAT * 4);

    { int n = Nn * FEAT;
      k_cast_bf16<<<(n + 255) / 256, 256, 0, stream>>>(x, actbf, n); }

    for (int l = 0; l < 4; l++) {
        const float* W  = (const float*)d_in[2 + 4 * l];
        const float* as = (const float*)d_in[3 + 4 * l];
        const float* ad = (const float*)d_in[4 + 4 * l];
        const float* b  = (const float*)d_in[5 + 4 * l];
        const int fin = FEAT, fout = HEADS * FEAT;   // fin == KDIM

        { int n = fin * fout;
          k_transpose_bf16<<<(n + 255) / 256, 256, 0, stream>>>(W, Wt, fin, fout); }
        { dim3 grid((fout + 127) / 128, (Nn + 63) / 64);
          k_wmma_gemm<<<grid, 256, 0, stream>>>(actbf, Wt, hbuf, Nn, fout, 0); }
        { int thr = Nn * HEADS * 32;
          k_logits<<<(thr + 255) / 256, 256, 0, stream>>>(hbuf, as, ad, al_s, al_d, Nn); }
        { int n = Nn * FEAT;
          k_init<<<(n + 255) / 256, 256, 0, stream>>>(mbuf, sbuf, agg, Nn * HEADS, n); }
        { int n = Etot * HEADS;
          k_edge_max<<<(n + 255) / 256, 256, 0, stream>>>(ei, al_s, al_d, mbuf, E, Etot); }
        { int n = Etot * HEADS;
          k_edge_exp<<<(n + 255) / 256, 256, 0, stream>>>(ei, al_s, al_d, mbuf, exbuf, sbuf, E, Etot); }
        { int blocks = (Etot + 7) / 8;
          k_aggregate<<<blocks, 256, 0, stream>>>(ei, hbuf, exbuf, sbuf, agg, E, Etot); }
        float* of = (l == 3) ? (out + (size_t)Nn * Nn) : actf;   // layer 3: z -> d_out tail
        { int n = Nn * FEAT;
          k_finalize<<<(n + 255) / 256, 256, 0, stream>>>(agg, b, of, actbf, Nn, (l == 3) ? 1 : 0); }
    }

    // adj = sigmoid(z @ z^T): A = z(bf16), Bt = z(bf16) -> B = z^T (K = FEAT = KDIM)
    { dim3 grid((Nn + 127) / 128, (Nn + 63) / 64);
      k_wmma_gemm<<<grid, 256, 0, stream>>>(actbf, actbf, out, Nn, Nn, 1); }
}
